// Model_84241488544507
// MI455X (gfx1250) — matrix operand order repaired and checked by reading, not compiled, not run on hardware
//
#include <hip/hip_runtime.h>
#include <hip/hip_bf16.h>

typedef __attribute__((ext_vector_type(16))) _Float16 v16h;
typedef __attribute__((ext_vector_type(8)))  _Float16 v8h;
typedef __attribute__((ext_vector_type(8)))  float    v8f;
typedef _Float16 h16;

#define A_DIM    20
#define L_DIM    512
#define CHN      8
#define NLAYERS  9
#define NTHREADS 256
#define NWAVES   8
#define HPAD     520   // padded row stride for h (1040 B, 16B-aligned) to spread banks

// ---- LDS layout (element offsets, _Float16 units) ----
#define OFF_SVT  0                          // V transposed [r][k], 32x32 (col-matmul B operand)
#define OFF_SW   (OFF_SVT + 32*32)          // conv weights [c][k], 16x32, k = t*8+ci
#define OFF_NZ   (OFF_SW + 16*32)           // nz one-hot f16 [512][32]
#define OFF_H    (OFF_NZ + 512*32)          // h [20][HPAD] f16
#define OFF_G0   (OFF_H  + A_DIM*HPAD)      // conv ping [a][p][ci], p<=256
#define OFF_G1   (OFF_G0 + A_DIM*256*CHN)   // conv pong [a][p][ci], p<=128
#define OFF_MB   (OFF_G1 + A_DIM*128*CHN)   // mask bits: 20*16 u32 (as 640 halves)
#define SMEM_HALVES (OFF_MB + A_DIM*16*2)
#define SMEM_BYTES  (SMEM_HALVES * 2)       // ~180.8 KB < 320 KB/WGP

__device__ __forceinline__ v8h lds_v8(const h16* p) { return *(const v8h*)p; }
__device__ __forceinline__ void sts_v8(h16* p, v8h v) { *(v8h*)p = v; }

__global__ __launch_bounds__(NTHREADS, 1)
void fused_msa_conv_kernel(const float* __restrict__ x,      // (B, 512, 20)
                           const float* __restrict__ lpm,    // (20, 20)
                           const float* __restrict__ pm,     // (20, 20)
                           const float* __restrict__ stdv,   // (1,)
                           const float* __restrict__ w0,     // (8, 1, 3)
                           const float* __restrict__ wsl,    // (8, 8, 8, 3)
                           float* __restrict__ out)          // (B, 20, 8)
{
  extern __shared__ h16 smem[];
  h16* sVT = smem + OFF_SVT;
  h16* sW  = smem + OFF_SW;
  h16* nz  = smem + OFF_NZ;
  h16* hH  = smem + OFF_H;
  h16* g0  = smem + OFF_G0;
  h16* g1  = smem + OFF_G1;
  unsigned int* mb = (unsigned int*)(smem + OFF_MB);   // mb[a*16 + w]

  const int b      = blockIdx.x;
  const int tid    = threadIdx.x;
  const int wid    = tid >> 5;
  const int lane   = tid & 31;
  const int lane16 = lane & 15;
  const bool hi    = lane >= 16;

  // ---- Phase 0: zero mask bitwords ----------------------------------------
  for (int idx = tid; idx < A_DIM * 16; idx += NTHREADS) mb[idx] = 0u;
  __syncthreads();

  // ---- Phase 1: V^T (f16 [r][k]) + load x -> nz f16 + mask bits -----------
  for (int idx = tid; idx < 32 * 32; idx += NTHREADS) {
    int r = idx >> 5, i = idx & 31;
    float v = 0.0f;
    if (i < A_DIM && r < A_DIM) {
      float Mir = fminf(fmaxf(lpm[i*A_DIM + r], 0.001f), 1.0f) * pm[i*A_DIM + r];
      float Mri = fminf(fmaxf(lpm[r*A_DIM + i], 0.001f), 1.0f) * pm[r*A_DIM + i];
      v = ((r > i) && (r <= A_DIM - 2) ? Mir : 0.0f) + (r < i ? Mri : 0.0f);
    }
    sVT[idx] = (h16)v;   // sVT[r*32 + k] = V[k][r]
  }
  // x row stride = 20 floats = 80 B -> float4 aligned (5 vec4 per position)
  const float4* xb = (const float4*)(x + (size_t)b * L_DIM * A_DIM);
  for (int idx = tid; idx < L_DIM * 5; idx += NTHREADS) {
    int s = idx / 5, c4 = idx - 5 * s;
    float4 v = xb[s * 5 + c4];
    int i0 = c4 * 4;
    nz[s*32 + i0 + 0] = (h16)v.x;
    nz[s*32 + i0 + 1] = (h16)v.y;
    nz[s*32 + i0 + 2] = (h16)v.z;
    nz[s*32 + i0 + 3] = (h16)v.w;
    // one-hot: at most one nonzero per s; set mask bit m[a][s]
    unsigned bit = 1u << (s & 31);
    if (v.x != 0.0f) atomicOr(&mb[(i0+0)*16 + (s>>5)], bit);
    if (v.y != 0.0f) atomicOr(&mb[(i0+1)*16 + (s>>5)], bit);
    if (v.z != 0.0f) atomicOr(&mb[(i0+2)*16 + (s>>5)], bit);
    if (v.w != 0.0f && i0 + 3 < A_DIM) atomicOr(&mb[(i0+3)*16 + (s>>5)], bit);
  }
  for (int idx = tid; idx < L_DIM * 12; idx += NTHREADS) {  // zero-pad i = 20..31
    int s = idx / 12, i = A_DIM + (idx - 12 * s);
    nz[s*32 + i] = (h16)0.0f;
  }
  __syncthreads();

  // ---- Phase 2a: column matmul via WMMA, write hH = acc -------------------
#pragma unroll 2
  for (int tile = wid; tile < 64; tile += NWAVES) {  // 32 s-tiles x 2 r-tiles
    int s0 = (tile >> 1) * 16, r0 = (tile & 1) * 16;
    int srow = s0 + lane16;
    // A (nz 16x32): av[0..7] = K (hi?8:0)+0..7 ; av[8..15] = K 16+(hi?8:0)+0..7
    v8h a_lo = lds_v8(&nz[srow*32 + (hi ? 8 : 0)]);
    v8h a_hi = lds_v8(&nz[srow*32 + 16 + (hi ? 8 : 0)]);
    // B (V 32x16): bv[0..7] = K (hi?8:0)+0..7 ; bv[8..15] = K 16+(hi?8:0)+0..7, column r0+lane16
    v8h b_lo = lds_v8(&sVT[(r0 + lane16)*32 + (hi ? 8 : 0)]);
    v8h b_hi = lds_v8(&sVT[(r0 + lane16)*32 + 16 + (hi ? 8 : 0)]);
    v16h av = __builtin_shufflevector(a_lo, a_hi, 0,1,2,3,4,5,6,7,8,9,10,11,12,13,14,15);
    v16h bv = __builtin_shufflevector(b_lo, b_hi, 0,1,2,3,4,5,6,7,8,9,10,11,12,13,14,15);
    v8f acc = {};
    acc = __builtin_amdgcn_wmma_f32_16x16x32_f16(false, av, false, bv,
                                                 (short)0, acc, false, false);
    // D: lane holds fixed r = r0+lane16, s = s0 + (hi?8:0) + v  (8 consecutive)
    int r = r0 + lane16;
    if (r < A_DIM) {
      v8h hv;
#pragma unroll
      for (int v = 0; v < 8; ++v) hv[v] = (h16)acc[v];
      sts_v8(&hH[r*HPAD + s0 + (hi ? 8 : 0)], hv);
    }
  }
  __syncthreads();

  // ---- Phase 2b: row term + mask term via bitmask words, RMW hH -----------
  const float sd = stdv[0];
  const float inv2 = __builtin_amdgcn_rcpf(sd * sd);   // 1/(std^2), one v_rcp_f32
  const float vk1 = __expf(-0.5f * inv2);
  const float vk2 = __expf(-2.0f * inv2);
  const float vk3 = __expf(-4.5f * inv2);

  for (int idx = tid; idx < A_DIM * 16; idx += NTHREADS) {
    int a = idx >> 4, w = idx & 15;
    unsigned cur  = mb[a*16 + w];
    unsigned prev = (w > 0)  ? mb[a*16 + w - 1] : 0u;
    unsigned next = (w < 15) ? mb[a*16 + w + 1] : 0u;
    unsigned mk[3];
#pragma unroll
    for (int k = 1; k <= 3; ++k) {
      unsigned sp = (cur << k) | (prev >> (32 - k));        // m[s-k]
      unsigned sn = (cur >> k) | (next << (32 - k));        // m[s+k]
      if (w == 15) {                                        // s = L-1: any(m[L-1-k:])
        unsigned topAny = (cur >> (31 - k)) != 0u;
        sp = (sp & 0x7FFFFFFFu) | (topAny << 31);
      }
      if (w == 0) {                                         // s = 0: any(m[0:k+1])
        unsigned lowAny = (cur & ((1u << (k + 1)) - 1u)) != 0u;
        sn = (sn & ~1u) | lowAny;
      }
      mk[k-1] = sp | sn;
    }
    unsigned sel3 = mk[2];
    unsigned sel2 = mk[1] & ~mk[2];
    unsigned sel1 = mk[0] & ~(mk[1] | mk[2]);
#pragma unroll
    for (int g = 0; g < 4; ++g) {
      h16* hp = &hH[a*HPAD + w*32 + g*8];
      v8h hv = lds_v8(hp);
#pragma unroll
      for (int j2 = 0; j2 < 8; ++j2) {
        int j = g*8 + j2;
        float add = ((cur  >> j) & 1u) ? 1.0f : 0.0f;       // xp/nz term (one-hot is 0/1)
        add += ((sel3 >> j) & 1u) ? vk3 :
               ((sel2 >> j) & 1u) ? vk2 :
               ((sel1 >> j) & 1u) ? vk1 : 0.0f;
        hv[j2] = (h16)((float)hv[j2] + add);
      }
      sts_v8(hp, hv);
    }
  }
  __syncthreads();

  // ---- Phase 3: layer 0 (1->8ch conv K=3) + pool, VALU; g0[a][q][ci] ------
  float w0r[CHN][3];
#pragma unroll
  for (int c = 0; c < CHN; ++c)
#pragma unroll
    for (int t = 0; t < 3; ++t) w0r[c][t] = w0[c*3 + t];

  for (int idx = tid; idx < A_DIM * 256; idx += NTHREADS) {
    int a = idx >> 8, q = idx & 255;
    int s0 = 2 * q;
    float hm1 = (s0 - 1 >= 0)    ? (float)hH[a*HPAD + s0 - 1] : 0.0f;
    float h0  =                    (float)hH[a*HPAD + s0];
    float h1  =                    (float)hH[a*HPAD + s0 + 1];
    float hp2 = (s0 + 2 < L_DIM) ? (float)hH[a*HPAD + s0 + 2] : 0.0f;
    v8h pv;
#pragma unroll
    for (int c = 0; c < CHN; ++c) {
      float y0 = w0r[c][0]*hm1 + w0r[c][1]*h0 + w0r[c][2]*h1;
      float y1 = w0r[c][0]*h0  + w0r[c][1]*h1 + w0r[c][2]*hp2;
      pv[c] = (h16)(0.5f * (y0 + y1));
    }
    sts_v8(&g0[(a*256 + q)*CHN], pv);
  }
  __syncthreads();

  // ---- Phase 4: layers 1..8, 8x8x3 conv via WMMA + pool -------------------
  // K-order k = t*8 + ci  =>  B operand = contiguous 8-channel position rows:
  //   lanes 0-15:  bv[0..7]=gin[a][p-1][:], bv[8..15]=gin[a][p][:]
  //   lanes 16-31: bv[0..7]=gin[a][p+1][:], bv[8..15]=0          (K 24..31 pad)
  h16* gin = g0;
  h16* gout = g1;
  int Lin = 256;
  for (int layer = 1; layer < NLAYERS; ++layer) {
    int Lout = Lin >> 1;
    const float* wl = wsl + (size_t)(layer - 1) * CHN * CHN * 3;
    for (int idx = tid; idx < 16 * 32; idx += NTHREADS) {
      int c = idx >> 5, k = idx & 31;
      float w = 0.0f;
      if (c < CHN && k < CHN * 3) {
        int t = k >> 3, ci = k & 7;
        w = wl[(c*CHN + ci)*3 + t];
      }
      sW[idx] = (h16)w;
    }
    __syncthreads();

    // weight A-operand (same for every tile this layer)
    v8h wa_lo = lds_v8(&sW[lane16*32 + (hi ? 16 : 0)]);
    v8h wa_hi = lds_v8(&sW[lane16*32 + (hi ? 16 : 0) + 8]);
    v16h av = __builtin_shufflevector(wa_lo, wa_hi, 0,1,2,3,4,5,6,7,8,9,10,11,12,13,14,15);

    int tpa = (Lin + 15) >> 4;
    int nt  = A_DIM * tpa;
#pragma unroll 2
    for (int tile = wid; tile < nt; tile += NWAVES) {
      int a = tile / tpa;
      int base = (tile - a * tpa) * 16;
      int p = base + lane16;
      v8h zero = {};
      int posA = hi ? p + 1 : p - 1;
      v8h b_lo = (p < Lin && posA >= 0 && posA < Lin)
                   ? lds_v8(&gin[(a*Lin + posA)*CHN]) : zero;
      v8h b_hi = (!hi && p < Lin)
                   ? lds_v8(&gin[(a*Lin + p)*CHN]) : zero;
      v16h bv = __builtin_shufflevector(b_lo, b_hi, 0,1,2,3,4,5,6,7,8,9,10,11,12,13,14,15);
      v8f acc = {};
      acc = __builtin_amdgcn_wmma_f32_16x16x32_f16(false, av, false, bv,
                                                   (short)0, acc, false, false);
      // lanes 0..15 hold c_out = v; pool adjacent positions via shfl_xor
      v8h pv;
#pragma unroll
      for (int v = 0; v < 8; ++v) {
        float y = acc[v];
        pv[v] = (h16)(0.5f * (y + __shfl_xor(y, 1, 32)));
      }
      if (!hi && ((lane16 & 1) == 0)) {
        int q = (base + lane16) >> 1;
        if (q < Lout) sts_v8(&gout[(a*Lout + q)*CHN], pv);
      }
    }
    __syncthreads();
    h16* tmp = gin; gin = gout; gout = tmp;
    Lin = Lout;
  }

  // ---- Phase 5: write (B, 20, 8) output -----------------------------------
  for (int idx = tid; idx < A_DIM * CHN; idx += NTHREADS) {
    out[(size_t)b * A_DIM * CHN + idx] = (float)gin[idx];
  }
}

extern "C" void kernel_launch(void* const* d_in, const int* in_sizes, int n_in,
                              void* d_out, int out_size, void* d_ws, size_t ws_size,
                              hipStream_t stream) {
  const float* x    = (const float*)d_in[0];
  // d_in[1] = masks: redundant (x is already one_hot * mask, values in {0,1})
  const float* lpm  = (const float*)d_in[2];
  const float* pm   = (const float*)d_in[3];
  const float* stdv = (const float*)d_in[4];
  const float* w0   = (const float*)d_in[5];
  const float* wsl  = (const float*)d_in[6];
  float* out = (float*)d_out;

  const int B = in_sizes[0] / (L_DIM * A_DIM);   // 1024
  fused_msa_conv_kernel<<<B, NTHREADS, SMEM_BYTES, stream>>>(
      x, lpm, pm, stdv, w0, wsl, out);
}
